// GRP_29566554866102
// MI455X (gfx1250) — matrix operand order repaired and checked by reading, not compiled, not run on hardware
//
#include <hip/hip_runtime.h>
#include <hip/hip_bf16.h>
#include <math.h>

typedef __bf16 bf16;
typedef __attribute__((ext_vector_type(16))) __bf16 v16bf;
typedef __attribute__((ext_vector_type(8)))  __bf16 v8bf;
typedef __attribute__((ext_vector_type(8)))  float  v8f;

__device__ __forceinline__ v8f wmma_bf16(v16bf a, v16bf b, v8f c) {
    // D = A(16x32 bf16) * B(32x16 bf16) + C(16x16 f32)
    return __builtin_amdgcn_wmma_f32_16x16x32_bf16(
        false, a, false, b, (short)0, c, false, false);
}
__device__ __forceinline__ v16bf cat8(v8bf lo, v8bf hi) {
    return __builtin_shufflevector(lo, hi, 0, 1, 2, 3, 4, 5, 6, 7,
                                   8, 9, 10, 11, 12, 13, 14, 15);
}

// ---------------------------------------------------------------------------
// Model dims
// ---------------------------------------------------------------------------
#define BATCH 32
#define IMGSZ 224
#define PSZ   16
#define DIM   768
#define NHEAD 12
#define HS    64
#define NLAY  12
#define MLPD  3072
#define NPATCH 196          // (224/16)^2
#define SEQ   426           // 1 + 196 + 1 + 196 + 32
#define TB    32
#define NTOK  (BATCH * SEQ) // 13632
#define AOUT  7

// ---------------------------------------------------------------------------
// Weight transpose+convert: out[n][k] = (bf16) in[k][n]; in is K x N fp32.
// LDS-tiled so both global accesses are coalesced.
// ---------------------------------------------------------------------------
__global__ __launch_bounds__(256)
void wconv_kernel(const float* __restrict__ in, bf16* __restrict__ out,
                  int K, int N)
{
    __shared__ float t[32][33];
    int k0 = blockIdx.y * 32, n0 = blockIdx.x * 32;
    int tx = threadIdx.x & 31, ty = threadIdx.x >> 5;
#pragma unroll
    for (int r = ty; r < 32; r += 8) {
        int gk = k0 + r, gn = n0 + tx;
        t[r][tx] = (gk < K && gn < N) ? in[(size_t)gk * N + gn] : 0.f;
    }
    __syncthreads();
#pragma unroll
    for (int r = ty; r < 32; r += 8) {
        int gn = n0 + r, gk = k0 + tx;
        if (gn < N && gk < K) out[(size_t)gn * K + gk] = (bf16)t[tx][r];
    }
}

// Head-layout variant for Wq/Wk/Wv (NH, D, HS): out[n=h*64+e][k] = in[h][k][e]
__global__ __launch_bounds__(256)
void wconv_perm_kernel(const float* __restrict__ in, bf16* __restrict__ out)
{
    __shared__ float t[32][33];
    int k0 = blockIdx.y * 32, n0 = blockIdx.x * 32;
    int tx = threadIdx.x & 31, ty = threadIdx.x >> 5;
#pragma unroll
    for (int r = ty; r < 32; r += 8) {
        int gk = k0 + r, gn = n0 + tx;
        t[r][tx] = in[(size_t)((gn >> 6) * DIM + gk) * HS + (gn & 63)];
    }
    __syncthreads();
#pragma unroll
    for (int r = ty; r < 32; r += 8)
        out[(size_t)(n0 + r) * DIM + k0 + tx] = (bf16)t[tx][r];
}

// ---------------------------------------------------------------------------
// bf16-WMMA GEMM: C[M,N] = act( A[M,K](bf16) @ Bt[N,K]^T(bf16) + bias + res )
// Block tile 128x128, K-step 64. 256 threads = 8 waves, wave tile 32x64.
// Interior tiles: register-pipelined staging (load next K-tile during WMMAs).
// ---------------------------------------------------------------------------
#define TM 128
#define TN 128
#define TK 64
#define LPITCH (TK + 8)      // 72 elems = 144 B rows (16B-aligned)

template<bool BIAS, bool RES, bool RELU, typename TC>
__global__ __launch_bounds__(256)
void gemm_kernel(const bf16* __restrict__ A, int lda,
                 const bf16* __restrict__ Bt, int ldb,   // Bt: N x K, row-major
                 TC* __restrict__ C, int ldc,
                 const float* __restrict__ bias,
                 const float* __restrict__ res, int ldr,
                 int M, int N, int K)
{
    __shared__ bf16 As[TM][LPITCH];
    __shared__ bf16 Bs[TN][LPITCH];   // Bs[n][k]

    const int tid  = threadIdx.x;
    const int lane = tid & 31;
    const int w    = tid >> 5;
    const int lh   = lane >> 4;      // lane half (0/1)
    const int lr   = lane & 15;
    const int m0   = blockIdx.y * TM;
    const int n0   = blockIdx.x * TN;
    const int wm   = (w & 3) * 32;   // wave M offset in tile
    const int wn   = (w >> 2) * 64;  // wave N offset in tile
    const bool interior = (m0 + TM <= M) && (n0 + TN <= N) &&
                          ((K & (TK - 1)) == 0);

    const int ar = tid >> 1, ac = (tid & 1) * 32;    // A staging coords
    const int bn = tid & 127, bk = (tid >> 7) * 32;  // B staging coords

    v8f acc[2][4];
#pragma unroll
    for (int mi = 0; mi < 2; ++mi)
#pragma unroll
        for (int ni = 0; ni < 4; ++ni)
#pragma unroll
            for (int j = 0; j < 8; ++j) acc[mi][ni][j] = 0.f;

    const bf16* ga = A  + (size_t)(m0 + ar) * lda + ac;
    const bf16* gb = Bt + (size_t)(n0 + bn) * ldb + bk;
    v8bf ra[4], rb[4];
    if (interior) {
#pragma unroll
        for (int t = 0; t < 4; ++t) {
            ra[t] = *(const v8bf*)(ga + 8 * t);
            rb[t] = *(const v8bf*)(gb + 8 * t);
        }
    }

    for (int kb = 0; kb < K; kb += TK) {
        if (interior) {
            // dump pipelined registers to LDS
#pragma unroll
            for (int t = 0; t < 4; ++t) {
                *(v8bf*)&As[ar][ac + 8 * t] = ra[t];
                *(v8bf*)&Bs[bn][bk + 8 * t] = rb[t];
            }
            __syncthreads();
            // issue next K-tile loads; they retire under the WMMAs below
            if (kb + TK < K) {
                ga += TK; gb += TK;
#pragma unroll
                for (int t = 0; t < 4; ++t) {
                    ra[t] = *(const v8bf*)(ga + 8 * t);
                    rb[t] = *(const v8bf*)(gb + 8 * t);
                }
            }
        } else {
            // guarded scalar staging (edge tiles)
            int gm = m0 + ar;
#pragma unroll
            for (int i = 0; i < 32; ++i) {
                int gk = kb + ac + i;
                As[ar][ac + i] = (gm < M && gk < K)
                                 ? A[(size_t)gm * lda + gk] : (bf16)0.f;
            }
            int gn = n0 + bn;
#pragma unroll
            for (int i = 0; i < 32; ++i) {
                int gk = kb + bk + i;
                Bs[bn][bk + i] = (gn < N && gk < K)
                                 ? Bt[(size_t)gn * ldb + gk] : (bf16)0.f;
            }
            __syncthreads();
        }

#pragma unroll
        for (int ks = 0; ks < TK; ks += 32) {
            v16bf af[2], bfg[4];
#pragma unroll
            for (int mi = 0; mi < 2; ++mi) {
                int r = wm + mi * 16 + lr;
                af[mi] = cat8(*(const v8bf*)&As[r][ks + lh * 8],
                              *(const v8bf*)&As[r][ks + 16 + lh * 8]);
            }
#pragma unroll
            for (int ni = 0; ni < 4; ++ni) {
                int cn = wn + ni * 16 + lr;
                bfg[ni] = cat8(*(const v8bf*)&Bs[cn][ks + lh * 8],
                               *(const v8bf*)&Bs[cn][ks + 16 + lh * 8]);
            }
#pragma unroll
            for (int mi = 0; mi < 2; ++mi)
#pragma unroll
                for (int ni = 0; ni < 4; ++ni)
                    acc[mi][ni] = wmma_bf16(af[mi], bfg[ni], acc[mi][ni]);
        }
        __syncthreads();
    }

    // epilogue: D layout -> row = half*8+j, col = lane&15
#pragma unroll
    for (int mi = 0; mi < 2; ++mi)
#pragma unroll
        for (int ni = 0; ni < 4; ++ni)
#pragma unroll
            for (int j = 0; j < 8; ++j) {
                int gm = m0 + wm + mi * 16 + lh * 8 + j;
                int gn = n0 + wn + ni * 16 + lr;
                if (interior || (gm < M && gn < N)) {
                    float v = acc[mi][ni][j];
                    if (BIAS) v += bias[gn];
                    if (RES)  v += res[(size_t)gm * ldr + gn];
                    if (RELU) v = v > 0.f ? v : 0.f;
                    C[(size_t)gm * ldc + gn] = (TC)v;
                }
            }
}

// ---------------------------------------------------------------------------
// Fused attention: one wave per (batch, head, 16-query tile).
// scores = (q @ k^T) * scale -> softmax -> @ v, all via bf16 WMMA.
// ---------------------------------------------------------------------------
#define NQT 27              // ceil(426/16)
#define NPAD 448            // 14 K-chunks of 32

__global__ __launch_bounds__(32)
void attn_kernel(const bf16* __restrict__ qkv, bf16* __restrict__ attout,
                 float scale)
{
    __shared__ float sc[16][NPAD];
    __shared__ bf16  wei[16][NPAD];
    __shared__ bf16  vT[HS][40];     // v chunk transposed, 80B rows

    const int lane = threadIdx.x;
    const int lh = lane >> 4, lr = lane & 15;
    const int qt = blockIdx.x % NQT;
    const int h  = (blockIdx.x / NQT) % NHEAD;
    const int b  = blockIdx.x / (NQT * NHEAD);
    const size_t base = (size_t)b * SEQ;
    const int q0 = qt * 16;

    // q fragments (16x64 as two 16x32 K-chunks) straight from global (b128s)
    v16bf aq[2];
    {
        int r = q0 + lr; if (r > SEQ - 1) r = SEQ - 1;
        const bf16* qp = qkv + (base + r) * (3 * DIM) + h * HS;
#pragma unroll
        for (int c = 0; c < 2; ++c)
            aq[c] = cat8(*(const v8bf*)(qp + c * 32 + lh * 8),
                         *(const v8bf*)(qp + c * 32 + 16 + lh * 8));
    }

    // scores: 27 tiles of 16 key columns
    for (int st = 0; st < NQT; ++st) {
        int s = st * 16 + lr; if (s > SEQ - 1) s = SEQ - 1;
        const bf16* kp = qkv + (base + s) * (3 * DIM) + DIM + h * HS;
        v8f a;
#pragma unroll
        for (int j = 0; j < 8; ++j) a[j] = 0.f;
#pragma unroll
        for (int c = 0; c < 2; ++c) {
            v16bf bk = cat8(*(const v8bf*)(kp + c * 32 + lh * 8),
                            *(const v8bf*)(kp + c * 32 + 16 + lh * 8));
            a = wmma_bf16(aq[c], bk, a);
        }
#pragma unroll
        for (int j = 0; j < 8; ++j)
            sc[lh * 8 + j][st * 16 + lr] = a[j] * scale;
    }
    __syncthreads();

    // mask padded key columns
    for (int s = SEQ + lane; s < NPAD; s += 32)
        for (int r = 0; r < 16; ++r) sc[r][s] = -1e30f;
    __syncthreads();

    // softmax per query row (wave32 reductions)
    for (int r = 0; r < 16; ++r) {
        float m = -1e30f;
#pragma unroll
        for (int s = 0; s < NPAD / 32; ++s) m = fmaxf(m, sc[r][lane + s * 32]);
#pragma unroll
        for (int o = 16; o; o >>= 1) m = fmaxf(m, __shfl_xor(m, o, 32));
        float sum = 0.f;
#pragma unroll
        for (int s = 0; s < NPAD / 32; ++s) {
            float e = __expf(sc[r][lane + s * 32] - m);
            sc[r][lane + s * 32] = e;
            sum += e;
        }
#pragma unroll
        for (int o = 16; o; o >>= 1) sum += __shfl_xor(sum, o, 32);
        float inv = 1.f / sum;
#pragma unroll
        for (int s = 0; s < NPAD / 32; ++s)
            wei[r][lane + s * 32] = (bf16)(sc[r][lane + s * 32] * inv);
    }
    __syncthreads();

    // att = wei(16x448) @ v(448x64)
    v8f acc[4];
#pragma unroll
    for (int ni = 0; ni < 4; ++ni)
#pragma unroll
        for (int j = 0; j < 8; ++j) acc[ni][j] = 0.f;

    for (int kc = 0; kc < NPAD / 32; ++kc) {
        int s = kc * 32 + lane;
        bool valid = s < SEQ;
        const bf16* vp = qkv + (base + (valid ? s : 0)) * (3 * DIM) +
                         2 * DIM + h * HS;
        v8bf vrow[8];
#pragma unroll
        for (int t = 0; t < 8; ++t) {
            if (valid) vrow[t] = *(const v8bf*)(vp + 8 * t);
            else
#pragma unroll
                for (int i = 0; i < 8; ++i) vrow[t][i] = (bf16)0.f;
        }
        __syncthreads();
#pragma unroll
        for (int j = 0; j < HS; ++j)
            vT[j][lane] = vrow[j >> 3][j & 7];
        __syncthreads();

        v16bf aw = cat8(*(const v8bf*)&wei[lr][kc * 32 + lh * 8],
                        *(const v8bf*)&wei[lr][kc * 32 + 16 + lh * 8]);
#pragma unroll
        for (int ni = 0; ni < 4; ++ni) {
            v16bf bv = cat8(*(const v8bf*)&vT[ni * 16 + lr][lh * 8],
                            *(const v8bf*)&vT[ni * 16 + lr][16 + lh * 8]);
            acc[ni] = wmma_bf16(aw, bv, acc[ni]);
        }
    }

#pragma unroll
    for (int ni = 0; ni < 4; ++ni)
#pragma unroll
        for (int j = 0; j < 8; ++j) {
            int r = q0 + lh * 8 + j;
            if (r < SEQ)
                attout[(base + r) * DIM + h * HS + ni * 16 + lr] =
                    (bf16)acc[ni][j];
        }
}

// ---------------------------------------------------------------------------
// LayerNorm over D=768. One 256-thread block per output row.
// row_in = blockIdx.x * row_step (row_step=426 extracts cls rows).
// ---------------------------------------------------------------------------
template<typename TOUT>
__global__ __launch_bounds__(256)
void ln_kernel(const float* __restrict__ x, int row_step,
               const float* __restrict__ g, const float* __restrict__ bb,
               TOUT* __restrict__ out)
{
    __shared__ float red[256];
    const int tid = threadIdx.x;
    const float* xr = x + (size_t)blockIdx.x * row_step * DIM;
    float v0 = xr[tid], v1 = xr[tid + 256], v2 = xr[tid + 512];
    red[tid] = v0 + v1 + v2;
    __syncthreads();
    for (int o = 128; o; o >>= 1) {
        if (tid < o) red[tid] += red[tid + o];
        __syncthreads();
    }
    float mean = red[0] * (1.f / DIM);
    __syncthreads();
    float d0 = v0 - mean, d1 = v1 - mean, d2 = v2 - mean;
    red[tid] = d0 * d0 + d1 * d1 + d2 * d2;
    __syncthreads();
    for (int o = 128; o; o >>= 1) {
        if (tid < o) red[tid] += red[tid + o];
        __syncthreads();
    }
    float rstd = rsqrtf(red[0] * (1.f / DIM) + 1e-5f);
    TOUT* orow = out + (size_t)blockIdx.x * DIM;
    orow[tid]       = (TOUT)(d0 * rstd * g[tid]       + bb[tid]);
    orow[tid + 256] = (TOUT)(d1 * rstd * g[tid + 256] + bb[tid + 256]);
    orow[tid + 512] = (TOUT)(d2 * rstd * g[tid + 512] + bb[tid + 512]);
}

// ---------------------------------------------------------------------------
// Patch extraction: (B,224,224,3) -> (B,196,768) bf16
// ---------------------------------------------------------------------------
__global__ __launch_bounds__(256)
void patches_kernel(const float* __restrict__ img, bf16* __restrict__ out)
{
    const int TOT = BATCH * NPATCH * DIM;
    int id = blockIdx.x * 256 + threadIdx.x;
    if (id >= TOT) return;
    int c = id % DIM;
    int p = (id / DIM) % NPATCH;
    int b = id / (DIM * NPATCH);
    int ph = p / 14, pw = p % 14;
    int py = c / 48, px = (c % 48) / 3, ch = c % 3;
    size_t pix = (((size_t)b * IMGSZ + ph * PSZ + py) * IMGSZ +
                  pw * PSZ + px) * 3 + ch;
    out[id] = (bf16)img[pix];
}

// ---------------------------------------------------------------------------
// Sequence assembly: x = concat(cls, obs, goal_tok, goal, lang) + pos_emb
// ---------------------------------------------------------------------------
__global__ __launch_bounds__(256)
void assemble_kernel(const float* __restrict__ pobs,
                     const float* __restrict__ pgoal,
                     const int* __restrict__ txt,
                     const float* __restrict__ tok_emb,
                     const float* __restrict__ pos_emb,
                     const float* __restrict__ cls_tok,
                     const float* __restrict__ goal_tok,
                     float* __restrict__ x)
{
    const int TOT = BATCH * SEQ * DIM;
    int id = blockIdx.x * 256 + threadIdx.x;
    if (id >= TOT) return;
    int d = id % DIM;
    int t = (id / DIM) % SEQ;
    int b = id / (DIM * SEQ);
    float v;
    if (t == 0)        v = cls_tok[d];
    else if (t <= 196) v = pobs[((size_t)b * NPATCH + (t - 1)) * DIM + d];
    else if (t == 197) v = goal_tok[d];
    else if (t <= 393) v = pgoal[((size_t)b * NPATCH + (t - 198)) * DIM + d];
    else               v = tok_emb[(size_t)txt[b * TB + (t - 394)] * DIM + d];
    x[id] = v + pos_emb[(size_t)t * DIM + d];
}

// ---------------------------------------------------------------------------
// Host launcher
// ---------------------------------------------------------------------------
extern "C" void kernel_launch(void* const* d_in, const int* in_sizes, int n_in,
                              void* d_out, int out_size, void* d_ws, size_t ws_size,
                              hipStream_t stream)
{
    const float* images   = (const float*)d_in[0];
    const float* goal_img = (const float*)d_in[1];
    const int*   txt      = (const int*)  d_in[2];
    const float* patch_W  = (const float*)d_in[3];
    const float* patch_b  = (const float*)d_in[4];
    const float* tok_emb  = (const float*)d_in[5];
    const float* pos_emb  = (const float*)d_in[6];
    const float* cls_tok  = (const float*)d_in[7];
    const float* goal_tok = (const float*)d_in[8];
    const float* Wq       = (const float*)d_in[9];
    const float* Wk       = (const float*)d_in[10];
    const float* Wv       = (const float*)d_in[11];
    const float* Wo       = (const float*)d_in[12];
    const float* bo       = (const float*)d_in[13];
    const float* ln1_g    = (const float*)d_in[14];
    const float* ln1_b    = (const float*)d_in[15];
    const float* ln2_g    = (const float*)d_in[16];
    const float* ln2_b    = (const float*)d_in[17];
    const float* W1       = (const float*)d_in[18];
    const float* b1       = (const float*)d_in[19];
    const float* W2       = (const float*)d_in[20];
    const float* b2       = (const float*)d_in[21];
    const float* lnf_g    = (const float*)d_in[22];
    const float* lnf_b    = (const float*)d_in[23];
    const float* ah_W1    = (const float*)d_in[24];
    const float* ah_b1    = (const float*)d_in[25];
    const float* ah_W2    = (const float*)d_in[26];
    const float* ah_b2    = (const float*)d_in[27];

    // workspace arena (bytes, all region sizes are multiples of 256)
    char* ws = (char*)d_ws;
    constexpr size_t SZ_X    = (size_t)NTOK * DIM * 4;       // fp32 residual
    constexpr size_t SZ_XN   = (size_t)NTOK * DIM * 2;       // bf16 LN out
    constexpr size_t SZ_QKV  = (size_t)NTOK * 3 * DIM * 2;   // bf16 q|k|v
    constexpr size_t SZ_ATT  = (size_t)NTOK * DIM * 2;       // bf16 attn out
    constexpr size_t SZ_H    = (size_t)NTOK * MLPD * 2;      // bf16 MLP hidden
    constexpr size_t OFF_X   = 0;
    constexpr size_t OFF_XN  = OFF_X + SZ_X;
    constexpr size_t OFF_QKV = OFF_XN + SZ_XN;
    constexpr size_t OFF_ATT = OFF_QKV + SZ_QKV;
    constexpr size_t OFF_H   = OFF_ATT + SZ_ATT;
    constexpr size_t OFF_WT  = OFF_H + SZ_H;   // bf16 transposed weight scratch

    float* x     = (float*)(ws + OFF_X);
    bf16*  xn    = (bf16*) (ws + OFF_XN);
    bf16*  qkv   = (bf16*) (ws + OFF_QKV);
    bf16*  att   = (bf16*) (ws + OFF_ATT);
    bf16*  hbuf  = (bf16*) (ws + OFF_H);
    bf16*  wt    = (bf16*) (ws + OFF_WT);      // up to 3072x768 bf16
    // overlapped pre/post-layer scratch:
    constexpr size_t SZ_PO = (size_t)BATCH * NPATCH * DIM * 4; // fp32 patch out
    float* pobs  = (float*)(ws + OFF_QKV);                  // inside qkv region
    float* pgoal = (float*)(ws + OFF_QKV + SZ_PO);
    constexpr size_t SZ_PA = (size_t)BATCH * NPATCH * DIM * 2; // bf16 patch pix
    bf16*  paobs  = (bf16*)(ws + OFF_H);                    // inside h region
    bf16*  pagoal = (bf16*)(ws + OFF_H + SZ_PA);
    bf16*  cls    = (bf16*)(ws + OFF_XN);                   // 32x768 bf16
    bf16*  hid    = (bf16*)(ws + OFF_H);                    // 32x3072 bf16
    (void)ws_size; (void)n_in; (void)in_sizes; (void)out_size;

    const dim3 blk(256);
    const dim3 gW24(DIM / 32, DIM / 32);        // 768x768 transpose
    const dim3 gW1t(MLPD / 32, DIM / 32);       // K=768 -> N=3072
    const dim3 gW2t(DIM / 32, MLPD / 32);       // K=3072 -> N=768
    const int PTOT = BATCH * NPATCH * DIM;

    // 1) patch pixel gather (bf16 A matrices)
    patches_kernel<<<(PTOT + 255) / 256, blk, 0, stream>>>(images,   paobs);
    patches_kernel<<<(PTOT + 255) / 256, blk, 0, stream>>>(goal_img, pagoal);

    // 2) patch embed GEMMs: (6272x768) @ (768x768) + patch_b
    {
        wconv_kernel<<<gW24, blk, 0, stream>>>(patch_W, wt, DIM, DIM);
        dim3 g((DIM + TN - 1) / TN, (BATCH * NPATCH + TM - 1) / TM);
        gemm_kernel<true, false, false, float><<<g, blk, 0, stream>>>(
            paobs, DIM, wt, DIM, pobs, DIM, patch_b, nullptr, 0,
            BATCH * NPATCH, DIM, DIM);
        gemm_kernel<true, false, false, float><<<g, blk, 0, stream>>>(
            pagoal, DIM, wt, DIM, pgoal, DIM, patch_b, nullptr, 0,
            BATCH * NPATCH, DIM, DIM);
    }

    // 3) assemble token stream + positional embedding
    {
        const int TOT = NTOK * DIM;
        assemble_kernel<<<(TOT + 255) / 256, blk, 0, stream>>>(
            pobs, pgoal, txt, tok_emb, pos_emb, cls_tok, goal_tok, x);
    }

    const float scale = 1.f / sqrtf((float)DIM);   // reference uses D^-0.5
    const dim3 gD((DIM + TN - 1) / TN, (NTOK + TM - 1) / TM);
    const dim3 gMLP((MLPD + TN - 1) / TN, (NTOK + TM - 1) / TM);

    // 4) transformer layers
    for (int l = 0; l < NLAY; ++l) {
        ln_kernel<bf16><<<NTOK, blk, 0, stream>>>(
            x, 1, ln1_g + l * DIM, ln1_b + l * DIM, xn);

        // q / k / v projections (head-layout weights -> transposed bf16)
        const size_t wofs = (size_t)l * NHEAD * DIM * HS;
        wconv_perm_kernel<<<gW24, blk, 0, stream>>>(Wq + wofs, wt);
        gemm_kernel<false, false, false, bf16><<<gD, blk, 0, stream>>>(
            xn, DIM, wt, DIM, qkv + 0 * DIM, 3 * DIM, nullptr, nullptr, 0,
            NTOK, DIM, DIM);
        wconv_perm_kernel<<<gW24, blk, 0, stream>>>(Wk + wofs, wt);
        gemm_kernel<false, false, false, bf16><<<gD, blk, 0, stream>>>(
            xn, DIM, wt, DIM, qkv + 1 * DIM, 3 * DIM, nullptr, nullptr, 0,
            NTOK, DIM, DIM);
        wconv_perm_kernel<<<gW24, blk, 0, stream>>>(Wv + wofs, wt);
        gemm_kernel<false, false, false, bf16><<<gD, blk, 0, stream>>>(
            xn, DIM, wt, DIM, qkv + 2 * DIM, 3 * DIM, nullptr, nullptr, 0,
            NTOK, DIM, DIM);

        attn_kernel<<<BATCH * NHEAD * NQT, dim3(32), 0, stream>>>(
            qkv, att, scale);

        // x = x + att @ Wo + bo
        wconv_kernel<<<gW24, blk, 0, stream>>>(
            Wo + (size_t)l * DIM * DIM, wt, DIM, DIM);
        gemm_kernel<true, true, false, float><<<gD, blk, 0, stream>>>(
            att, DIM, wt, DIM, x, DIM, bo + l * DIM, x, DIM, NTOK, DIM, DIM);

        ln_kernel<bf16><<<NTOK, blk, 0, stream>>>(
            x, 1, ln2_g + l * DIM, ln2_b + l * DIM, xn);

        // h = relu(xn @ W1 + b1)
        wconv_kernel<<<gW1t, blk, 0, stream>>>(
            W1 + (size_t)l * DIM * MLPD, wt, DIM, MLPD);
        gemm_kernel<true, false, true, bf16><<<gMLP, blk, 0, stream>>>(
            xn, DIM, wt, DIM, hbuf, MLPD, b1 + l * MLPD, nullptr, 0,
            NTOK, MLPD, DIM);

        // x = x + h @ W2 + b2
        wconv_kernel<<<gW2t, blk, 0, stream>>>(
            W2 + (size_t)l * MLPD * DIM, wt, MLPD, DIM);
        gemm_kernel<true, true, false, float><<<gD, blk, 0, stream>>>(
            hbuf, MLPD, wt, MLPD, x, DIM, b2 + l * DIM, x, DIM,
            NTOK, DIM, MLPD);
    }

    // 5) final LN on cls rows only (row t=0 of each batch)
    ln_kernel<bf16><<<BATCH, blk, 0, stream>>>(x, SEQ, lnf_g, lnf_b, cls);

    // 6) action head: relu(cls @ ah_W1 + ah_b1) @ ah_W2 + ah_b2
    {
        wconv_kernel<<<gW1t, blk, 0, stream>>>(ah_W1, wt, DIM, MLPD);
        dim3 g1((MLPD + TN - 1) / TN, (BATCH + TM - 1) / TM);
        gemm_kernel<true, false, true, bf16><<<g1, blk, 0, stream>>>(
            cls, DIM, wt, DIM, hid, MLPD, ah_b1, nullptr, 0,
            BATCH, MLPD, DIM);
        dim3 gw((AOUT + 31) / 32, MLPD / 32);
        wconv_kernel<<<gw, blk, 0, stream>>>(ah_W2, wt, MLPD, AOUT);
        dim3 g2((AOUT + TN - 1) / TN, (BATCH + TM - 1) / TM);
        gemm_kernel<true, false, false, float><<<g2, blk, 0, stream>>>(
            hid, MLPD, wt, MLPD, (float*)d_out, AOUT, ah_b2, nullptr, 0,
            BATCH, AOUT, MLPD);
    }
}
